// SNN_84765474554279
// MI455X (gfx1250) — matrix operand order repaired and checked by reading, not compiled, not run on hardware
//
#include <hip/hip_runtime.h>

// ---------------------------------------------------------------------------
// SNN MLP + LIF for MI455X (gfx1250, wave32, WMMA bf16 3x-split)
//   h   = x @ W1^T + b1     [8192, 512]   (d_ws)
//   cur = h @ W2^T + b2     [8192, 2048]  (d_out)
//   spikes = LIF_scan(cur)  in place in d_out
// LDS tiles are stored pre-swizzled into the WMMA register layout so the
// inner loop is ds_load_b128 + v_wmma only; bf16 split happens once per
// block tile during staging.
// ---------------------------------------------------------------------------

typedef __attribute__((ext_vector_type(16))) __bf16 v16bf;
typedef __attribute__((ext_vector_type(2)))  __bf16 bf16x2;
typedef __attribute__((ext_vector_type(8)))  float  v8f;
typedef __attribute__((ext_vector_type(2)))  float  f32x2;
typedef __attribute__((ext_vector_type(4)))  float  f32x4;

#define BM 128
#define BN 128
#define BK 32
#define NTA (BM / 16)      // 8 A-fragment tiles per block tile
#define NTB (BN / 16)      // 8 B-fragment tiles per block tile
#define TELEM 512          // 32 lanes * 16 bf16 per fragment tile

// --- fragment element offsets (ISA 7.12.2 layouts); even-k pairs land on
// --- consecutive j so staging can do packed bf16x2 stores ---------------
__device__ __forceinline__ int a_frag_off(int m, int k) {
    // A 16x32: lane = m + ((k&15)>=8 ? 16:0); j = (k>=16?8:0) + (k&7)
    int lane = m + (((k & 15) >= 8) ? 16 : 0);
    int j = ((k & 16) >> 1) + (k & 7);
    return lane * 16 + j;
}
__device__ __forceinline__ int b_frag_off(int n, int k) {
    // B 32x16: lane = n + ((k&15)>=8 ? 16:0); j = (k>=16?8:0) + (k&7)
    int lane = n + (((k & 15) >= 8) ? 16 : 0);
    int j = ((k & 16) >> 1) + (k & 7);
    return lane * 16 + j;
}

// stage one BMxBK (or BNxBK) f32 tile -> bf16 hi/lo in WMMA frag layout
__device__ __forceinline__ void stage_a(const float* __restrict__ A, int K,
                                        int row0, int k0, int tid,
                                        __bf16* sh, __bf16* sl) {
    #pragma unroll
    for (int i = 0; i < 8; ++i) {
        int p = tid + i * 256;            // pair index, 2048 pairs
        int r = p >> 4;                   // 16 pairs per 32-float row
        int c = (p & 15) * 2;             // even k within tile
        f32x2 v = *(const f32x2*)(A + (size_t)(row0 + r) * K + k0 + c);
        __bf16 h0 = (__bf16)v.x, h1 = (__bf16)v.y;
        __bf16 l0 = (__bf16)(v.x - (float)h0);
        __bf16 l1 = (__bf16)(v.y - (float)h1);
        int off = (r >> 4) * TELEM + a_frag_off(r & 15, c);
        bf16x2 hp; hp[0] = h0; hp[1] = h1;
        bf16x2 lp; lp[0] = l0; lp[1] = l1;
        *(bf16x2*)(sh + off) = hp;
        *(bf16x2*)(sl + off) = lp;
    }
}
__device__ __forceinline__ void stage_b(const float* __restrict__ W, int K,
                                        int col0, int k0, int tid,
                                        __bf16* sh, __bf16* sl) {
    #pragma unroll
    for (int i = 0; i < 8; ++i) {
        int p = tid + i * 256;
        int r = p >> 4;
        int c = (p & 15) * 2;
        f32x2 v = *(const f32x2*)(W + (size_t)(col0 + r) * K + k0 + c);
        __bf16 h0 = (__bf16)v.x, h1 = (__bf16)v.y;
        __bf16 l0 = (__bf16)(v.x - (float)h0);
        __bf16 l1 = (__bf16)(v.y - (float)h1);
        int off = (r >> 4) * TELEM + b_frag_off(r & 15, c);
        bf16x2 hp; hp[0] = h0; hp[1] = h1;
        bf16x2 lp; lp[0] = l0; lp[1] = l1;
        *(bf16x2*)(sh + off) = hp;
        *(bf16x2*)(sl + off) = lp;
    }
}

__global__ __launch_bounds__(256) void gemm_tn_bf16x3(
    const float* __restrict__ A,    // [M, K]
    const float* __restrict__ W,    // [N, K]   C = A * W^T
    const float* __restrict__ bias, // [N]
    float* __restrict__ C,          // [M, N]
    int M, int N, int K)
{
    __shared__ __align__(32) __bf16 sAhi[2][NTA * TELEM];
    __shared__ __align__(32) __bf16 sAlo[2][NTA * TELEM];
    __shared__ __align__(32) __bf16 sBhi[2][NTB * TELEM];
    __shared__ __align__(32) __bf16 sBlo[2][NTB * TELEM];

    const int tid  = threadIdx.x;
    const int lane = tid & 31;
    const int wid  = tid >> 5;       // 8 waves: 2 (rows) x 4 (cols)
    const int wm   = wid & 1;        // wave tile = 64 rows x 32 cols
    const int wn   = wid >> 1;

    const int row0 = blockIdx.x * BM;
    const int col0 = blockIdx.y * BN;
    const int lhalf = lane >> 4;
    const int l15   = lane & 15;

    const v8f vzero = {0.f, 0.f, 0.f, 0.f, 0.f, 0.f, 0.f, 0.f};
    v8f acc[4][2];
    #pragma unroll
    for (int i = 0; i < 4; ++i)
        #pragma unroll
        for (int j = 0; j < 2; ++j)
            acc[i][j] = vzero;

    const int nk = K / BK;

    stage_a(A, K, row0, 0, tid, sAhi[0], sAlo[0]);
    stage_b(W, K, col0, 0, tid, sBhi[0], sBlo[0]);
    __syncthreads();

    for (int kt = 0; kt < nk; ++kt) {
        const int cur = kt & 1;
        if (kt + 1 < nk) {
            stage_a(A, K, row0, (kt + 1) * BK, tid, sAhi[cur ^ 1], sAlo[cur ^ 1]);
            stage_b(W, K, col0, (kt + 1) * BK, tid, sBhi[cur ^ 1], sBlo[cur ^ 1]);
        }
        if (kt + 2 < nk) {   // global_prefetch_b8 for the tile after next
            __builtin_prefetch(A + (size_t)(row0 + (tid >> 1)) * K + (kt + 2) * BK, 0, 1);
            __builtin_prefetch(W + (size_t)(col0 + (tid >> 1)) * K + (kt + 2) * BK, 0, 1);
        }

        const __bf16* Ah = sAhi[cur];
        const __bf16* Al = sAlo[cur];
        const __bf16* Bh = sBhi[cur];
        const __bf16* Bl = sBlo[cur];

        // fragment fetch: 2x ds_load_b128 per v16bf (lane-contiguous 32B)
        v16bf ah[4], al[4], bh[2], bl[2];
        #pragma unroll
        for (int mi = 0; mi < 4; ++mi) {
            int t = (wm * 4 + mi) * TELEM + lane * 16;
            ah[mi] = *(const v16bf*)(Ah + t);
            al[mi] = *(const v16bf*)(Al + t);
        }
        #pragma unroll
        for (int ni = 0; ni < 2; ++ni) {
            int t = (wn * 2 + ni) * TELEM + lane * 16;
            bh[ni] = *(const v16bf*)(Bh + t);
            bl[ni] = *(const v16bf*)(Bl + t);
        }

        // 3x-bf16 split: acc += hi*hi + lo*hi + hi*lo   (24 WMMA / k-step)
        #pragma unroll
        for (int mi = 0; mi < 4; ++mi) {
            #pragma unroll
            for (int ni = 0; ni < 2; ++ni) {
                acc[mi][ni] = __builtin_amdgcn_wmma_f32_16x16x32_bf16(
                    false, ah[mi], false, bh[ni], (short)0, acc[mi][ni], false, false);
                acc[mi][ni] = __builtin_amdgcn_wmma_f32_16x16x32_bf16(
                    false, al[mi], false, bh[ni], (short)0, acc[mi][ni], false, false);
                acc[mi][ni] = __builtin_amdgcn_wmma_f32_16x16x32_bf16(
                    false, ah[mi], false, bl[ni], (short)0, acc[mi][ni], false, false);
            }
        }
        __syncthreads();
    }

    // epilogue: D layout — VGPR r: lanes0-15 (M=r, N=lane), lanes16-31 (M=r+8)
    #pragma unroll
    for (int ni = 0; ni < 2; ++ni) {
        int n = col0 + wn * 32 + ni * 16 + l15;
        float bv = bias[n];
        #pragma unroll
        for (int mi = 0; mi < 4; ++mi) {
            #pragma unroll
            for (int r = 0; r < 8; ++r) {
                int m = row0 + wm * 64 + mi * 16 + r + lhalf * 8;
                C[(size_t)m * N + n] = acc[mi][ni][r] + bv;
            }
        }
    }
}

// LIF scan: tau=2, v_th=1, hard reset. One thread owns 4 (b,f) columns for
// all T; reads cur then overwrites with spikes in place (column-private).
__global__ __launch_bounds__(256) void lif_scan(float* __restrict__ io,
                                                int plane4, int T)
{
    int idx = blockIdx.x * 256 + threadIdx.x;
    if (idx >= plane4) return;
    f32x4* p = (f32x4*)io + idx;
    f32x4 v = {0.f, 0.f, 0.f, 0.f};
    for (int t = 0; t < T; ++t) {
        f32x4 c = p[(size_t)t * plane4];
        f32x4 s;
        #pragma unroll
        for (int e = 0; e < 4; ++e) {
            float hm = v[e] + (c[e] - v[e]) * 0.5f;   // decay_input, tau=2
            float sp = (hm >= 1.0f) ? 1.0f : 0.0f;    // Heaviside, v_th=1
            v[e] = hm * (1.0f - sp);                  // hard reset to 0
            s[e] = sp;
        }
        p[(size_t)t * plane4] = s;
    }
}

extern "C" void kernel_launch(void* const* d_in, const int* in_sizes, int n_in,
                              void* d_out, int out_size, void* d_ws, size_t ws_size,
                              hipStream_t stream) {
    const float* x  = (const float*)d_in[0];   // [T,B,F]
    const float* W1 = (const float*)d_in[1];   // [H,F]
    const float* b1 = (const float*)d_in[2];   // [H]
    const float* W2 = (const float*)d_in[3];   // [F,H]
    const float* b2 = (const float*)d_in[4];   // [F]
    float* out = (float*)d_out;                // [T,B,F]
    float* h   = (float*)d_ws;                 // [T*B, H] scratch

    const int T = 64, B = 128, F = 2048, H = F / 4;   // H = 512
    const int M = T * B;                               // 8192

    dim3 blk(256);

    // h = x @ W1^T + b1   (M=8192, N=512, K=2048)
    dim3 g1(M / BM, H / BN);
    gemm_tn_bf16x3<<<g1, blk, 0, stream>>>(x, W1, b1, h, M, H, F);

    // cur = h @ W2^T + b2 (M=8192, N=2048, K=512) -> d_out
    dim3 g2(M / BM, F / BN);
    gemm_tn_bf16x3<<<g2, blk, 0, stream>>>(h, W2, b2, out, M, F, H);

    // LIF scan in place (float4 per thread)
    int plane4 = (B * F) / 4;                          // 65536
    lif_scan<<<(plane4 + 255) / 256, blk, 0, stream>>>(out, plane4, T);
}
